// DLRM_Net_7636451852642
// MI455X (gfx1250) — matrix operand order repaired and checked by reading, not compiled, not run on hardware
//
#include <hip/hip_runtime.h>
#include <hip/hip_bf16.h>

typedef __attribute__((ext_vector_type(16))) __bf16 v16bf;
typedef __attribute__((ext_vector_type(8)))  float  v8f;

constexpr int Bn = 16384;
constexpr int Ln = 4;
constexpr int Dn = 64;
constexpr int Tn = 26;
constexpr long long Vn = 200000;
constexpr int RWp = 416;   // top MLP input width 415, padded to 416 (K%32==0)

__device__ __forceinline__ unsigned short f2bf(float f) {
    unsigned int u = __float_as_uint(f);
    unsigned int r = u + 0x7FFFu + ((u >> 16) & 1u);   // round-to-nearest-even
    return (unsigned short)(r >> 16);
}
__device__ __forceinline__ float bf2f(unsigned short s) {
    return __uint_as_float(((unsigned int)s) << 16);
}
// Low 32 bits of a generic pointer to LDS == LDS byte offset (ISA 10.2 aperture rule)
__device__ __forceinline__ unsigned lds_off(const void* p) {
    return (unsigned)(size_t)p;
}
__device__ __forceinline__ void async_load_b128(unsigned lds, const void* gptr) {
    unsigned long long ga = (unsigned long long)gptr;
    asm volatile("global_load_async_to_lds_b128 %0, %1, off"
                 :: "v"(lds), "v"(ga) : "memory");
}
__device__ __forceinline__ void wait_async0() {
    asm volatile("s_wait_asynccnt 0x0" ::: "memory");
}
// in-order completion: <=2 outstanding means the previous stage's 2 loads landed
__device__ __forceinline__ void wait_async2() {
    asm volatile("s_wait_asynccnt 0x2" ::: "memory");
}

// ---------------------------------------------------------------------------
// fp32 -> bf16 with column padding: dst[M x Kp] = src[M x K], pad cols = 0
// ---------------------------------------------------------------------------
__global__ __launch_bounds__(256) void convert_pad_kernel(
    const float* __restrict__ src, unsigned short* __restrict__ dst,
    int M, int K, int Kp)
{
    int i = blockIdx.x * 256 + threadIdx.x;
    if (i >= M * Kp) return;
    int r = i / Kp, c = i - r * Kp;
    dst[i] = (c < K) ? f2bf(src[(size_t)r * K + c]) : (unsigned short)0;
}

// ---------------------------------------------------------------------------
// WMMA GEMM:  C[M,N](bf16) = act( A[M,Kp](bf16) * W[N,Kp](bf16)^T + bias[N] )
// Kp % 32 == 0, buffers pre-padded -> no predication anywhere.
// Block: 256 threads (8 waves), 64x64 block tile, K chunks of 32.
// Double-buffered LDS staging via global_load_async_to_lds_b128:
// stage k+1 is in flight (wait asynccnt<=2) while stage k feeds the WMMAs.
// Wave w computes 16x16 tiles (mt=w>>2, nt=w&3) and (mt+2, nt).
// ---------------------------------------------------------------------------
__global__ __launch_bounds__(256) void gemm_bias_relu(
    const unsigned short* __restrict__ A, const unsigned short* __restrict__ W,
    const float* __restrict__ bias, unsigned short* __restrict__ C,
    int M, int N, int Kp, int do_relu)
{
    __shared__ unsigned short As[2][64 * 32];   // 2 x 4KB
    __shared__ unsigned short Ws[2][64 * 32];

    const int t    = threadIdx.x;
    const int lane = t & 31;
    const int wave = t >> 5;
    const int mBase = blockIdx.y * 64;
    const int nBase = blockIdx.x * 64;

    const int nt  = wave & 3;
    const int mt0 = wave >> 2;   // 0..1
    const int mt1 = mt0 + 2;     // 2..3

    v8f acc0 = {0.f, 0.f, 0.f, 0.f, 0.f, 0.f, 0.f, 0.f};
    v8f acc1 = acc0;

    const int rlo   = lane & 15;
    const int khalf = lane >> 4;

    // this thread's 16B staging chunk: row = t>>2, cols [(t&3)*8, +8)
    const int srow = t >> 2;
    const int scol = (t & 3) * 8;
    const unsigned ldsA0 = lds_off(&As[0][0]) + (unsigned)t * 16u;
    const unsigned ldsW0 = lds_off(&Ws[0][0]) + (unsigned)t * 16u;
    const unsigned short* gA = A + (size_t)(mBase + srow) * Kp + scol;
    const unsigned short* gW = W + (size_t)(nBase + srow) * Kp + scol;

    const int nChunks = Kp >> 5;

    // prologue: stage chunk 0 into buffer 0
    async_load_b128(ldsA0, gA);
    async_load_b128(ldsW0, gW);
    gA += 32;
    gW += 32;

    for (int kc = 0; kc < nChunks; ++kc) {
        const int cur = kc & 1;
        const bool more = (kc + 1) < nChunks;
        if (more) {
            const unsigned boff = (unsigned)(1 - cur) * (64 * 32 * 2);
            async_load_b128(ldsA0 + boff, gA);
            async_load_b128(ldsW0 + boff, gW);
            gA += 32;
            gW += 32;
            wait_async2();     // chunk kc landed; kc+1 still in flight
        } else {
            wait_async0();
        }
        __syncthreads();

        // fragments in contiguous k order (k = 16h + s), same for A and B
        const unsigned short* Ac = As[cur];
        const unsigned short* Wc = Ws[cur];
        alignas(32) unsigned short af0[16], af1[16], bfr[16];
        #pragma unroll
        for (int j = 0; j < 8; ++j) {
            int ka = khalf * 16 + j;       // K 16h..16h+7
            int kb = khalf * 16 + 8 + j;   // K 16h+8..16h+15
            af0[j]     = Ac[(mt0 * 16 + rlo) * 32 + ka];
            af0[j + 8] = Ac[(mt0 * 16 + rlo) * 32 + kb];
            af1[j]     = Ac[(mt1 * 16 + rlo) * 32 + ka];
            af1[j + 8] = Ac[(mt1 * 16 + rlo) * 32 + kb];
        }
        #pragma unroll
        for (int j = 0; j < 16; ++j) {
            bfr[j] = Wc[(nt * 16 + rlo) * 32 + khalf * 16 + j];
        }
        v16bf a0, a1, bb;
        __builtin_memcpy(&a0, af0, 32);
        __builtin_memcpy(&a1, af1, 32);
        __builtin_memcpy(&bb, bfr, 32);

        acc0 = __builtin_amdgcn_wmma_f32_16x16x32_bf16(false, a0, false, bb, (short)0, acc0, false, false);
        acc1 = __builtin_amdgcn_wmma_f32_16x16x32_bf16(false, a1, false, bb, (short)0, acc1, false, false);
        __syncthreads();   // all reads of buf[cur] retired before it is refilled
    }

    // epilogue: bias + activation, bf16 store
    const int col   = nBase + nt * 16 + rlo;
    const int mhalf = khalf * 8;
    const float bv  = bias[col];
    #pragma unroll
    for (int r = 0; r < 8; ++r) {
        float v0 = acc0[r] + bv;
        float v1 = acc1[r] + bv;
        if (do_relu) { v0 = v0 > 0.f ? v0 : 0.f; v1 = v1 > 0.f ? v1 : 0.f; }
        C[(size_t)(mBase + mt0 * 16 + mhalf + r) * N + col] = f2bf(v0);
        C[(size_t)(mBase + mt1 * 16 + mhalf + r) * N + col] = f2bf(v1);
    }
}

// ---------------------------------------------------------------------------
// Embedding bag-sum: Tcat[b][1+t][d] = sum_{l<4} emb[t][ lS_i[t][b*4+l] ][d]
// Tcat padded to 32 rows per sample (rows 27..31 left uninitialized).
// ---------------------------------------------------------------------------
__global__ void embed_bag_kernel(const long long* __restrict__ lS_i,
                                 const float* __restrict__ emb,
                                 unsigned short* __restrict__ Tcat)
{
    const int d = threadIdx.x;     // 0..63
    const int b = blockIdx.x;
    const int t = blockIdx.y;
    const long long* idx = lS_i + (size_t)t * (Bn * Ln) + (size_t)b * Ln;
    float s = 0.f;
    #pragma unroll
    for (int l = 0; l < Ln; ++l) {
        long long ix = idx[l];
        s += emb[((size_t)t * Vn + (size_t)ix) * Dn + d];
    }
    Tcat[((size_t)b * 32 + 1 + t) * Dn + d] = f2bf(s);
}

// ---------------------------------------------------------------------------
// x (bf16) -> Tcat row 0 and R[:, 0:64]; also zero the R padding column 415.
// ---------------------------------------------------------------------------
__global__ void xcopy_kernel(const unsigned short* __restrict__ x,
                             unsigned short* __restrict__ Tcat,
                             unsigned short* __restrict__ R)
{
    const int d = threadIdx.x;
    const int b = blockIdx.x;
    unsigned short v = x[(size_t)b * Dn + d];
    Tcat[((size_t)b * 32) * Dn + d] = v;
    R[(size_t)b * RWp + d] = v;
    if (d == 0) R[(size_t)b * RWp + 415] = 0;   // K padding for top GEMM
}

// ---------------------------------------------------------------------------
// Interaction: per sample b, Z = Tcat·Tcatᵀ (27x27, K=64) via WMMA;
// strictly-lower-tri 351 entries -> R[b][64..414] (bf16).
// 4 waves/block, one sample per wave. Stage 32x64 bf16 tile (4KB) via
// async b128 loads (8 chunks per lane). Rows 27..31 hold garbage but every
// retained output depends only on rows/cols < 27, so it's inert.
// Tiles (0,0),(1,0),(1,1) only — tile (0,1) has no lower-tri entries.
// ---------------------------------------------------------------------------
__global__ __launch_bounds__(128) void interact_kernel(
    const unsigned short* __restrict__ Tcat,  // B x 32 x 64 bf16 (padded rows)
    unsigned short* __restrict__ R)           // B x 416 bf16
{
    __shared__ unsigned short Tl[4][32 * 64];
    const int t    = threadIdx.x;
    const int lane = t & 31;
    const int wv   = t >> 5;
    const int b    = blockIdx.x * 4 + wv;

    const char* gbase = (const char*)(Tcat + (size_t)b * 32 * Dn);
    const unsigned lbase = lds_off(&Tl[wv][0]);
    #pragma unroll
    for (int i = 0; i < 8; ++i) {
        const int c = i * 32 + lane;      // 16B chunk id, 256 chunks = 4KB
        async_load_b128(lbase + (unsigned)c * 16u, gbase + (size_t)c * 16u);
    }
    wait_async0();
    __syncthreads();

    const unsigned short* Ts = Tl[wv];
    const int rlo   = lane & 15;
    const int khalf = lane >> 4;

    const int mts[3] = {0, 1, 1};
    const int nts[3] = {0, 0, 1};
    #pragma unroll
    for (int ti = 0; ti < 3; ++ti) {
        const int mt = mts[ti], nt = nts[ti];
        v8f acc = {0.f, 0.f, 0.f, 0.f, 0.f, 0.f, 0.f, 0.f};
        #pragma unroll
        for (int kc = 0; kc < 2; ++kc) {
            alignas(32) unsigned short af[16], bfr[16];
            #pragma unroll
            for (int j = 0; j < 8; ++j) {
                int ka = kc * 32 + khalf * 16 + j;
                int kb = kc * 32 + khalf * 16 + 8 + j;
                af[j]     = Ts[(mt * 16 + rlo) * 64 + ka];
                af[j + 8] = Ts[(mt * 16 + rlo) * 64 + kb];
            }
            #pragma unroll
            for (int j = 0; j < 16; ++j) {
                bfr[j] = Ts[(nt * 16 + rlo) * 64 + kc * 32 + khalf * 16 + j];
            }
            v16bf a, bb;
            __builtin_memcpy(&a, af, 32);
            __builtin_memcpy(&bb, bfr, 32);
            acc = __builtin_amdgcn_wmma_f32_16x16x32_bf16(false, a, false, bb, (short)0, acc, false, false);
        }
        #pragma unroll
        for (int r = 0; r < 8; ++r) {
            int i = mt * 16 + khalf * 8 + r;
            int j = nt * 16 + rlo;
            if (i < Tn + 1 && j < i) {
                R[(size_t)b * RWp + Dn + (i * (i - 1)) / 2 + j] = f2bf(acc[r]);
            }
        }
    }
}

// ---------------------------------------------------------------------------
// Final layer: out[b] = sigmoid( act[b,:256](bf16) . Wt2[0,:256] + bt2[0] )
// ---------------------------------------------------------------------------
__global__ __launch_bounds__(256) void top_final_kernel(
    const unsigned short* __restrict__ act, const float* __restrict__ Wt2,
    const float* __restrict__ bt2, float* __restrict__ out)
{
    const int b = blockIdx.x * 256 + threadIdx.x;
    float s = bt2[0];
    const unsigned short* a = act + (size_t)b * 256;
    #pragma unroll 8
    for (int k = 0; k < 256; ++k) s += bf2f(a[k]) * Wt2[k];
    out[b] = 1.f / (1.f + __expf(-s));
}

// ---------------------------------------------------------------------------
extern "C" void kernel_launch(void* const* d_in, const int* in_sizes, int n_in,
                              void* d_out, int out_size, void* d_ws, size_t ws_size,
                              hipStream_t stream)
{
    const float*     dense_x = (const float*)d_in[0];
    /* d_in[1] = lS_o: offsets are exactly b*L, unused */
    const long long* lS_i    = (const long long*)d_in[2];
    const float*     emb     = (const float*)d_in[3];
    const float *Wb0 = (const float*)d_in[4],  *bb0 = (const float*)d_in[5];
    const float *Wb1 = (const float*)d_in[6],  *bb1 = (const float*)d_in[7];
    const float *Wb2 = (const float*)d_in[8],  *bb2 = (const float*)d_in[9];
    const float *Wt0 = (const float*)d_in[10], *bt0 = (const float*)d_in[11];
    const float *Wt1 = (const float*)d_in[12], *bt1 = (const float*)d_in[13];
    const float *Wt2 = (const float*)d_in[14], *bt2 = (const float*)d_in[15];

    char*  ws  = (char*)d_ws;
    size_t off = 0;
    auto alloc = [&](size_t bytes) -> void* {
        void* p = ws + off;
        off += (bytes + 255) & ~(size_t)255;
        return p;
    };
    typedef unsigned short u16;
    u16* Xp    = (u16*)alloc((size_t)Bn * 32  * 2);   // dense_x, K 13->32
    u16* Wb0p  = (u16*)alloc((size_t)512 * 32  * 2);
    u16* Wb1p  = (u16*)alloc((size_t)256 * 512 * 2);
    u16* Wb2p  = (u16*)alloc((size_t)64  * 256 * 2);
    u16* Wt0p  = (u16*)alloc((size_t)512 * RWp * 2);  // K 415->416
    u16* Wt1p  = (u16*)alloc((size_t)256 * 512 * 2);
    u16* act512 = (u16*)alloc((size_t)Bn * 512 * 2);
    u16* act256 = (u16*)alloc((size_t)Bn * 256 * 2);
    u16* xbuf   = (u16*)alloc((size_t)Bn * Dn  * 2);
    u16* Rbuf   = (u16*)alloc((size_t)Bn * RWp * 2);
    u16* Tcat   = (u16*)alloc((size_t)Bn * 32 * Dn * 2);   // 32 rows/sample

    const dim3 blk(256);
    auto cvt = [&](const float* s, u16* d, int M, int K, int Kp) {
        int n = M * Kp;
        convert_pad_kernel<<<dim3((n + 255) / 256), blk, 0, stream>>>(s, d, M, K, Kp);
    };

    // one-time (per launch) bf16 conversion + K padding
    cvt(dense_x, Xp,   Bn, 13, 32);
    cvt(Wb0,    Wb0p, 512, 13, 32);
    cvt(Wb1,    Wb1p, 256, 512, 512);
    cvt(Wb2,    Wb2p,  64, 256, 256);
    cvt(Wt0,    Wt0p, 512, 415, RWp);
    cvt(Wt1,    Wt1p, 256, 512, 512);

    // Bottom MLP: 13 -> 512 -> 256 -> 64 (ReLU)
    gemm_bias_relu<<<dim3(512 / 64, Bn / 64), blk, 0, stream>>>(Xp,     Wb0p, bb0, act512, Bn, 512, 32,  1);
    gemm_bias_relu<<<dim3(256 / 64, Bn / 64), blk, 0, stream>>>(act512, Wb1p, bb1, act256, Bn, 256, 512, 1);
    gemm_bias_relu<<<dim3( 64 / 64, Bn / 64), blk, 0, stream>>>(act256, Wb2p, bb2, xbuf,   Bn,  64, 256, 1);

    // x -> Tcat row 0 + R[:, :64]; zero R pad column
    xcopy_kernel<<<dim3(Bn), dim3(Dn), 0, stream>>>(xbuf, Tcat, Rbuf);

    // Embedding bag sums -> Tcat rows 1..26
    embed_bag_kernel<<<dim3(Bn, Tn), dim3(Dn), 0, stream>>>(lS_i, emb, Tcat);

    // Pairwise interactions -> R[:, 64:415]
    interact_kernel<<<dim3(Bn / 4), dim3(128), 0, stream>>>(Tcat, Rbuf);

    // Top MLP: 415(+pad) -> 512 -> 256 (ReLU), 256 -> 1 (sigmoid)
    gemm_bias_relu<<<dim3(512 / 64, Bn / 64), blk, 0, stream>>>(Rbuf,   Wt0p, bt0, act512, Bn, 512, RWp, 1);
    gemm_bias_relu<<<dim3(256 / 64, Bn / 64), blk, 0, stream>>>(act512, Wt1p, bt1, act256, Bn, 256, 512, 1);
    top_final_kernel<<<dim3(Bn / 256), dim3(256), 0, stream>>>(act256, Wt2, bt2, (float*)d_out);
}
